// AscendQwen3_5GatedDeltaNet_65060164600178
// MI455X (gfx1250) — matrix operand order repaired and checked by reading, not compiled, not run on hardware
//
#include <hip/hip_runtime.h>
#include <hip/hip_bf16.h>
#include <cstdint>
#include <cstddef>

// ---------------- problem constants ----------------
#define B_SZ   2
#define T_SZ   2048
#define HKq    16
#define HVv    32
#define DKk    128
#define DVv    128
#define CONVD  8192      // 2*HK*DK + HV*DV
#define KW     4
#define BC     64        // chunk length (matmul M/K granularity)
#define NC     (T_SZ / BC)
#define DVH    64        // DV split per block
#define NSPLIT 2         // DV / DVH
#define NTHR   256       // 8 wave32s
#define WROWS  (BC + KW - 1)   // 67-row conv window

typedef __attribute__((ext_vector_type(16))) __bf16 v16bf;
typedef __attribute__((ext_vector_type(8)))  __bf16 v8bf;
typedef __attribute__((ext_vector_type(8)))  float  v8f;
typedef __attribute__((ext_vector_type(4)))  float  v4f;
typedef __attribute__((ext_vector_type(4)))  unsigned int v4u;
typedef __attribute__((ext_vector_type(8)))  int v8i;
typedef __attribute__((ext_vector_type(4)))  int v4i;

#if __has_builtin(__builtin_amdgcn_tensor_load_to_lds) && \
    __has_builtin(__builtin_amdgcn_s_wait_tensorcnt)
#define HAVE_TDM 1
#else
#define HAVE_TDM 0
#endif

// ---------------- LDS layout (dynamic, ~259 KB of 320 KB/WGP) ----------------
constexpr int OFF_ST  = 0;         // f32  [64][128]  S^T
constexpr int OFF_SBT = 32768;     // bf16 [64][128]  S^T (B-operand, n-major)
constexpr int OFF_K   = 49152;     // bf16 [64][128]  k rows (A-op; n-major view of K^T)
constexpr int OFF_KT  = 65536;     // bf16 [128][64]  k^T (B-op for S update)
constexpr int OFF_Q   = 81920;     // bf16 [64][128]  q rows (A-operand)
constexpr int OFF_A   = 98304;     // f32  [64][64]   A (UT transform)
constexpr int OFF_U   = 114688;    // f32  [64][64]   V -> pseudo-V -> U
constexpr int OFF_PB  = 131072;    // bf16 [64][64]   P (A-operand, row-major)
constexpr int OFF_UBT = 139264;    // bf16 [64][64]   U^T (B-operand, n-major)
constexpr int OFF_UCT = 147456;    // bf16 [64][64]   (c_s*U)^T (A-op for S update)
constexpr int OFF_QS  = 155648;    // f32  [64][64]   e^G * Q@S0 (output C-init)
constexpr int OFF_G   = 172032;    // f32  [64]       cumulative gate G
constexpr int OFF_BE  = 172288;    // f32  [64]       beta
constexpr int OFF_CS  = 172544;    // f32  [64]       c_s = beta_s * e^(G_C - G_s)
constexpr int OFF_XQ  = 172800;    // f32  [67][128]  conv window, q slice (TDM dest)
constexpr int OFF_XK  = 207104;    // f32  [67][128]  conv window, k slice
constexpr int OFF_XV  = 241408;    // f32  [67][64]   conv window, v slice
constexpr int OFF_W   = 258560;    // f32  [4][320]   conv weights, tap-major
constexpr int OFF_BI  = 263680;    // f32  [320]      conv bias
constexpr int SMEM_BYTES = 264960;

// ---------------- WMMA helpers (CDNA5 bf16 16x16x32, f32 accum) ----------------
__device__ __forceinline__ v8f wmma_bf16(v16bf a, v16bf b, v8f c) {
  return __builtin_amdgcn_wmma_f32_16x16x32_bf16(false, a, false, b, (short)0, c,
                                                 false, false);
}
__device__ __forceinline__ v16bf load_a(const __bf16* base, int ld, int m0, int k0,
                                        int lane) {
  v16bf a;
  const int m = m0 + (lane & 15);
  const int kh = (lane < 16) ? 0 : 16;
#pragma unroll
  for (int e = 0; e < 16; ++e) {
    int k = k0 + kh + e;
    a[e] = base[m * ld + k];
  }
  return a;
}
__device__ __forceinline__ v16bf load_b_nmajor(const __bf16* base, int ld, int k0,
                                               int n0, int lane) {
  v16bf b;
  const int n = n0 + (lane & 15);
  const int kb = k0 + ((lane < 16) ? 0 : 16);
#pragma unroll
  for (int e = 0; e < 16; ++e) b[e] = base[n * ld + kb + e];
  return b;
}
__device__ __forceinline__ v8f load_c(const float* base, int ld, int m0, int n0,
                                      int lane) {
  v8f c;
  const int n = n0 + (lane & 15);
  const int mh = (lane < 16) ? 0 : 8;
#pragma unroll
  for (int j = 0; j < 8; ++j) c[j] = base[(m0 + mh + j) * ld + n];
  return c;
}

// ---------------- TDM: 2D tile (width x rows, f32) global -> LDS ----------------
#if HAVE_TDM
__device__ __forceinline__ void tdm_load_2d(const float* gaddr, unsigned lds_off,
                                            int width, int rows, int stride_elems) {
  unsigned long long ga = (unsigned long long)(uintptr_t)gaddr;
  v4u g0;
  g0[0] = 1u;                                            // count=1 (user D#)
  g0[1] = lds_off;                                       // LDS byte address
  g0[2] = (unsigned)(ga & 0xFFFFFFFFu);                  // global_addr[31:0]
  g0[3] = (unsigned)((ga >> 32) & 0x01FFFFFFu) | (2u << 30);  // addr[56:32]|type=2
  v8i g1;
  g1[0] = (int)(2u << 16);                               // data_size = 4 bytes
  g1[1] = (int)((unsigned)width << 16);                  // tensor_dim0[15:0]
  g1[2] = (int)((unsigned)rows << 16);                   // dim0 hi=0 | tensor_dim1 lo
  g1[3] = (int)((unsigned)width << 16);                  // dim1 hi=0 | tile_dim0
  g1[4] = rows;                                          // tile_dim1 (tile_dim2=0)
  g1[5] = stride_elems;                                  // tensor_dim0_stride lo32
  g1[6] = 0;                                             // stride hi | dim1_stride lo
  g1[7] = 0;
  v4i z4 = {0, 0, 0, 0};
#if defined(__clang_major__) && (__clang_major__ >= 23)
  v8i z8 = {0, 0, 0, 0, 0, 0, 0, 0};
  __builtin_amdgcn_tensor_load_to_lds(g0, g1, z4, z4, z8, 0);
#else
  __builtin_amdgcn_tensor_load_to_lds(g0, g1, z4, z4, 0);
#endif
}
#endif

// issue the 3 slice loads for the chunk whose first timestep is t0
__device__ __forceinline__ void stage_window_issue(const float* xb, int t0,
                                                   int qch0, int kch0, int vch0) {
#if HAVE_TDM
  const int tlo = t0 - (KW - 1);
  const int rows   = (tlo < 0) ? BC : WROWS;
  const int rowoff = (tlo < 0) ? (KW - 1) : 0;
  const int ts     = (tlo < 0) ? 0 : tlo;
  const float* g = xb + (size_t)ts * CONVD;
  tdm_load_2d(g + qch0, OFF_XQ + rowoff * DKk * 4, DKk, rows, CONVD);
  tdm_load_2d(g + kch0, OFF_XK + rowoff * DKk * 4, DKk, rows, CONVD);
  tdm_load_2d(g + vch0, OFF_XV + rowoff * DVH * 4, DVH, rows, CONVD);
#endif
}

// fallback: cooperative synchronous copy (used only if TDM builtins are absent)
__device__ __forceinline__ void stage_window_copy(char* smem, const float* xb, int t0,
                                                  int qch0, int kch0, int vch0,
                                                  int tid) {
  float* sXq = (float*)(smem + OFF_XQ);
  float* sXk = (float*)(smem + OFF_XK);
  float* sXv = (float*)(smem + OFF_XV);
  const int tlo = t0 - (KW - 1);
  const int rows   = (tlo < 0) ? BC : WROWS;
  const int rowoff = (tlo < 0) ? (KW - 1) : 0;
  const int ts     = (tlo < 0) ? 0 : tlo;
  for (int idx = tid; idx < rows * 80; idx += NTHR) {  // 80 float4 per row
    int r = idx / 80, sg = idx % 80;
    const float* gp;
    float* lp;
    if (sg < 32)      { gp = xb + (size_t)(ts + r) * CONVD + qch0 + sg * 4;
                        lp = sXq + (rowoff + r) * DKk + sg * 4; }
    else if (sg < 64) { gp = xb + (size_t)(ts + r) * CONVD + kch0 + (sg - 32) * 4;
                        lp = sXk + (rowoff + r) * DKk + (sg - 32) * 4; }
    else              { gp = xb + (size_t)(ts + r) * CONVD + vch0 + (sg - 64) * 4;
                        lp = sXv + (rowoff + r) * DVH + (sg - 64) * 4; }
    *(v4f*)lp = *(const v4f*)gp;
  }
}

// depthwise causal conv (K taps) + SiLU, from the staged LDS window
template <int NCH>
__device__ __forceinline__ void conv_taps(const float* __restrict__ sXs, int ldx,
                                          int col0, int row,
                                          const float* __restrict__ sWt,
                                          const float* __restrict__ sBias,
                                          int wch0, float* vals) {
#pragma unroll
  for (int j = 0; j < NCH; ++j) vals[j] = sBias[wch0 + j];
#pragma unroll
  for (int jt = 0; jt < KW; ++jt) {
    const float* xr = sXs + (row + jt) * ldx + col0;   // contiguous
    const float* wr = sWt + jt * 320 + wch0;           // contiguous (tap-major)
#pragma unroll
    for (int j = 0; j < NCH; ++j) vals[j] = fmaf(wr[j], xr[j], vals[j]);
  }
#pragma unroll
  for (int j = 0; j < NCH; ++j) vals[j] = vals[j] / (1.f + __expf(-vals[j]));
}

__global__ __launch_bounds__(NTHR, 1)
void gdn_fwd_kernel(const float* __restrict__ mixed, const float* __restrict__ a_in,
                    const float* __restrict__ b_in, const float* __restrict__ convw,
                    const float* __restrict__ convb, const float* __restrict__ A_log,
                    const float* __restrict__ dt_bias, float* __restrict__ out) {
  extern __shared__ char smem[];
  float*  sST  = (float*)(smem + OFF_ST);
  __bf16* sSbT = (__bf16*)(smem + OFF_SBT);
  __bf16* sK   = (__bf16*)(smem + OFF_K);
  __bf16* sKt  = (__bf16*)(smem + OFF_KT);
  __bf16* sQ   = (__bf16*)(smem + OFF_Q);
  float*  sA   = (float*)(smem + OFF_A);
  float*  sU   = (float*)(smem + OFF_U);
  __bf16* sPb  = (__bf16*)(smem + OFF_PB);
  __bf16* sUbT = (__bf16*)(smem + OFF_UBT);
  __bf16* sUcT = (__bf16*)(smem + OFF_UCT);
  float*  sQS  = (float*)(smem + OFF_QS);
  float*  sG   = (float*)(smem + OFF_G);
  float*  sBe  = (float*)(smem + OFF_BE);
  float*  sCs  = (float*)(smem + OFF_CS);
  float*  sXq  = (float*)(smem + OFF_XQ);
  float*  sXk  = (float*)(smem + OFF_XK);
  float*  sXv  = (float*)(smem + OFF_XV);
  float*  sWt  = (float*)(smem + OFF_W);
  float*  sBias= (float*)(smem + OFF_BI);

  const int tid  = threadIdx.x;
  const int lane = tid & 31;
  const int wv   = tid >> 5;
  const int dvh  = blockIdx.x;   // DV half
  const int h    = blockIdx.y;   // v-head
  const int bb   = blockIdx.z;   // batch
  const int hk   = h >> 1;       // GVA: q/k head = h / (HV/HK)

  const float* xb = mixed + (size_t)bb * T_SZ * CONVD;
  const int qch0 = hk * DKk;
  const int kch0 = HKq * DKk + hk * DKk;
  const int vch0 = 2 * HKq * DKk + h * DVv + dvh * DVH;
  const float negExpA = -__expf(A_log[h]);
  const float dtb = dt_bias[h];

  // ---- prologue: zero state, zero conv left-pad rows, stage weights (once) ----
  for (int i = tid; i < DVH * DKk; i += NTHR) { sST[i] = 0.f; sSbT[i] = (__bf16)0.f; }
  for (int i = tid; i < (KW - 1) * DKk; i += NTHR) { sXq[i] = 0.f; sXk[i] = 0.f; }
  for (int i = tid; i < (KW - 1) * DVH; i += NTHR) sXv[i] = 0.f;
  for (int i = tid; i < 320; i += NTHR) {
    int gch = (i < 128) ? (qch0 + i) : (i < 256) ? (kch0 + i - 128) : (vch0 + i - 256);
    sBias[i] = convb[gch];
#pragma unroll
    for (int jt = 0; jt < KW; ++jt) sWt[jt * 320 + i] = convw[gch * KW + jt];
  }
  __syncthreads();
#if HAVE_TDM
  if (wv == 0) {   // TDM: async DMA of chunk-0 window, wave0 issues + waits
    stage_window_issue(xb, 0, qch0, kch0, vch0);
    __builtin_amdgcn_s_wait_tensorcnt(0);
  }
  __syncthreads();
#else
  stage_window_copy(smem, xb, 0, qch0, kch0, vch0, tid);
  __syncthreads();
#endif

  for (int ci = 0; ci < NC; ++ci) {
    const int t0 = ci * BC;

#if HAVE_TDM
    if (ci > 0) {  // wait for the window DMA issued during the previous chunk
      if (wv == 0) __builtin_amdgcn_s_wait_tensorcnt(0);
      __syncthreads();
    }
#else
    if (ci > 0) { stage_window_copy(smem, xb, t0, qch0, kch0, vch0, tid);
                  __syncthreads(); }
#endif

    // ---- prefetch chunk ci+2's window into L2 ahead of its TDM issue ----
    if (ci + 2 < NC) {
      const int tn0 = t0 + 2 * BC;
      for (int idx = tid; idx < 640; idx += NTHR) {
        int row = idx / 10, seg = idx % 10;
        int ch = (seg < 4) ? (qch0 + seg * 32)
               : (seg < 8) ? (kch0 + (seg - 4) * 32)
                           : (vch0 + (seg - 8) * 32);
        __builtin_prefetch(&xb[(size_t)(tn0 + row) * CONVD + ch], 0, 3);
      }
    }

    // ---- phase 1: gating + conv/SiLU/l2norm from LDS window ----
    if (tid < BC) {
      int t = t0 + tid;
      float av = a_in[((size_t)bb * T_SZ + t) * HVv + h] + dtb;
      float sp = (av > 20.f) ? av : log1pf(__expf(av));
      sG[tid] = negExpA * sp;  // raw g_t (prefix-summed below)
      float bv = b_in[((size_t)bb * T_SZ + t) * HVv + h];
      sBe[tid] = 1.f / (1.f + __expf(-bv));
    }
    {
      const int row = tid >> 2, sub = tid & 3, c0 = sub * 32;
      float vals[32];
      // q
      conv_taps<32>(sXq, DKk, c0, row, sWt, sBias, c0, vals);
      float ss = 0.f;
#pragma unroll
      for (int j = 0; j < 32; ++j) ss = fmaf(vals[j], vals[j], ss);
      ss += __shfl_xor(ss, 1); ss += __shfl_xor(ss, 2);
      float inv = rsqrtf(ss + 1e-6f) * 0.08838834764831845f;
#pragma unroll
      for (int g2 = 0; g2 < 4; ++g2) {
        v8bf pk;
#pragma unroll
        for (int j = 0; j < 8; ++j) pk[j] = (__bf16)(vals[g2 * 8 + j] * inv);
        *(v8bf*)(sQ + row * DKk + c0 + g2 * 8) = pk;
      }
      // k (row-major + transposed)
      conv_taps<32>(sXk, DKk, c0, row, sWt, sBias, 128 + c0, vals);
      ss = 0.f;
#pragma unroll
      for (int j = 0; j < 32; ++j) ss = fmaf(vals[j], vals[j], ss);
      ss += __shfl_xor(ss, 1); ss += __shfl_xor(ss, 2);
      inv = rsqrtf(ss + 1e-6f);
#pragma unroll
      for (int g2 = 0; g2 < 4; ++g2) {
        v8bf pk;
#pragma unroll
        for (int j = 0; j < 8; ++j) {
          __bf16 kb = (__bf16)(vals[g2 * 8 + j] * inv);
          pk[j] = kb;
          sKt[(c0 + g2 * 8 + j) * BC + row] = kb;   // transpose scatter (once)
        }
        *(v8bf*)(sK + row * DKk + c0 + g2 * 8) = pk;
      }
      // v
      float vv[16];
      conv_taps<16>(sXv, DVH, sub * 16, row, sWt, sBias, 256 + sub * 16, vv);
#pragma unroll
      for (int j = 0; j < 16; ++j) sU[row * DVH + sub * 16 + j] = vv[j];
    }
    __syncthreads();
    if (tid == 0) {  // 64-element serial prefix sum (negligible)
      float acc = 0.f;
      for (int i = 0; i < BC; ++i) { acc += sG[i]; sG[i] = acc; }
    }
#if HAVE_TDM
    // window consumed: overlap next chunk's DMA with the rest of this chunk
    if (wv == 0 && ci + 1 < NC) stage_window_issue(xb, t0 + BC, qch0, kch0, vch0);
#endif
    __syncthreads();
    if (tid < BC) sCs[tid] = sBe[tid] * __expf(sG[BC - 1] - sG[tid]);

    // ---- phase 2: WMMA K@K^T, Q@K^T, K@S0, Q@S0 + fused in-register epilogue ----
#pragma unroll
    for (int i2 = 0; i2 < 2; ++i2) {
      const int ti = wv * 2 + i2;                      // 16 tiles / 8 waves
      const int m0 = (ti >> 2) * 16, n0 = (ti & 3) * 16;
      v8f accA = {}, accP = {}, accKS = {}, accQS = {};
#pragma unroll
      for (int kb = 0; kb < 4; ++kb) {
        v16bf aK  = load_a(sK, DKk, m0, kb * 32, lane);
        v16bf aQ  = load_a(sQ, DKk, m0, kb * 32, lane);
        v16bf bKt = load_b_nmajor(sK, DKk, kb * 32, n0, lane);    // K^T (contig)
        v16bf bS  = load_b_nmajor(sSbT, DKk, kb * 32, n0, lane);  // S0 (contig)
        accA  = wmma_bf16(aK, bKt, accA);
        accP  = wmma_bf16(aQ, bKt, accP);
        accKS = wmma_bf16(aK, bS,  accKS);
        accQS = wmma_bf16(aQ, bS,  accQS);
      }
      const int r = n0 + (lane & 15);    // column (s for A/P, n for KS/QS)
      const int mh = (lane < 16) ? 0 : 8;
      const float gr = sG[r], br = sBe[r];
#pragma unroll
      for (int j = 0; j < 8; ++j) {
        const int m = m0 + mh + j;       // row (t index)
        const float gm = sG[m];
        const float lam = __expf(gm);
        const float sc = (r <= m) ? __expf(gm - gr) * br : 0.f;
        sA[m * BC + r]  = (r < m) ? sc * accA[j] : 0.f;            // strict-lower A
        sPb[m * BC + r] = (__bf16)((r <= m) ? sc * accP[j] : 0.f); // causal P (bf16)
        sU[m * DVH + r] = fmaf(-lam, accKS[j], sU[m * DVH + r]);   // pseudo-values
        sQS[m * DVH + r] = lam * accQS[j];                         // e^G * Q@S0
      }
    }
    __syncthreads();

    // ---- phase 3: (I+A) U = Vp, forward substitution, 4 lanes per DV column ----
    {
      const int n = tid >> 2, q4 = tid & 3;  // 4 wave-contiguous lanes per column
      for (int s = 1; s < BC; ++s) {
        float val = sU[s * DVH + n];
        float part = 0.f;
        for (int r = q4; r < s; r += 4)
          part = fmaf(sA[s * BC + r], sU[r * DVH + n], part);
        part += __shfl_xor(part, 1);
        part += __shfl_xor(part, 2);
        sU[s * DVH + n] = val - part;          // 4 lanes write identical value
      }
    }
    __syncthreads();
    {  // U -> bf16 transposed (n-major) + c_s-scaled copy for the state update
      const int i0 = tid * 16;
      const int s = i0 >> 6, nb = i0 & 63;
      const float cs = sCs[s];
#pragma unroll
      for (int j = 0; j < 16; ++j) {
        float u = sU[i0 + j];
        int n = nb + j;
        sUbT[n * BC + s] = (__bf16)u;
        sUcT[n * BC + s] = (__bf16)(u * cs);
      }
    }
    __syncthreads();

    // ---- phase 4: O = e^G Q S0 + P @ U; write to global ----
#pragma unroll
    for (int i2 = 0; i2 < 2; ++i2) {
      const int ti = wv * 2 + i2;
      const int m0 = (ti >> 2) * 16, n0 = (ti & 3) * 16;
      v8f acc = load_c(sQS, DVH, m0, n0, lane);
#pragma unroll
      for (int kb = 0; kb < 2; ++kb)
        acc = wmma_bf16(load_a(sPb, BC, m0, kb * 32, lane),
                        load_b_nmajor(sUbT, BC, kb * 32, n0, lane), acc);
      const int n = n0 + (lane & 15);
      const int mh = (lane < 16) ? 0 : 8;
#pragma unroll
      for (int j = 0; j < 8; ++j) {
        int t = t0 + m0 + mh + j;
        out[((size_t)bb * T_SZ + t) * (HVv * DVv) + h * DVv + dvh * DVH + n] = acc[j];
      }
    }
    // no barrier: phase 5 reads only phase-1/3 data and per-wave-owned state tiles

    // ---- phase 5: S^T = e^(G_C) S^T + (c.U)^T @ K  (64x128, K=64) ----
    {
      const float lamC = __expf(sG[BC - 1]);
#pragma unroll
      for (int i2 = 0; i2 < 4; ++i2) {
        const int ti = wv * 4 + i2;            // 32 tiles: 4 (dv) x 8 (dk)
        const int m0 = (ti & 3) * 16, n0 = (ti >> 2) * 16;
        const int n = n0 + (lane & 15);        // dk index
        const int mh = (lane < 16) ? 0 : 8;
        v8f acc;
#pragma unroll
        for (int j = 0; j < 8; ++j) acc[j] = lamC * sST[(m0 + mh + j) * DKk + n];
#pragma unroll
        for (int kb = 0; kb < 2; ++kb)
          acc = wmma_bf16(load_a(sUcT, BC, m0, kb * 32, lane),        // (c.U)^T
                          load_b_nmajor(sKt, BC, kb * 32, n0, lane),  // K (contig)
                          acc);
#pragma unroll
        for (int j = 0; j < 8; ++j) {
          int m = m0 + mh + j;
          sST[m * DKk + n]  = acc[j];
          sSbT[m * DKk + n] = (__bf16)acc[j];
        }
      }
    }
    __syncthreads();
  }
}

extern "C" void kernel_launch(void* const* d_in, const int* in_sizes, int n_in,
                              void* d_out, int out_size, void* d_ws, size_t ws_size,
                              hipStream_t stream) {
  (void)in_sizes; (void)n_in; (void)d_ws; (void)ws_size; (void)out_size;
  const float* mixed = (const float*)d_in[0];
  const float* a_in  = (const float*)d_in[1];
  const float* b_in  = (const float*)d_in[2];
  const float* convw = (const float*)d_in[3];
  const float* convb = (const float*)d_in[4];
  const float* A_log = (const float*)d_in[5];
  const float* dtb   = (const float*)d_in[6];
  float* out = (float*)d_out;

  (void)hipFuncSetAttribute((const void*)gdn_fwd_kernel,
                            hipFuncAttributeMaxDynamicSharedMemorySize, SMEM_BYTES);
  dim3 grid(NSPLIT, HVv, B_SZ);   // 128 blocks: (dv-half, v-head, batch)
  gdn_fwd_kernel<<<grid, NTHR, SMEM_BYTES, stream>>>(mixed, a_in, b_in, convw,
                                                     convb, A_log, dtb, out);
}
